// MoEFFN_14173392077091
// MI455X (gfx1250) — matrix operand order repaired and checked by reading, not compiled, not run on hardware
//
#include <hip/hip_runtime.h>

#define D_DIM 1024
#define F_DIM 1024
#define E_NUM 8
#define BM 128
#define BN 128
#define BK 32
#define TS_STR 40   // padded LDS row stride (ushorts): 80B, 16B-aligned, spreads banks

#if defined(__has_builtin)
#if __has_builtin(__builtin_amdgcn_global_load_async_to_lds_b128)
#define ASYNC_LDS 1
#endif
#endif

typedef __bf16 v16bf __attribute__((ext_vector_type(16)));
typedef float v8f __attribute__((ext_vector_type(8)));
typedef unsigned short v8us __attribute__((ext_vector_type(8)));
typedef unsigned short v4us __attribute__((ext_vector_type(4)));

// GCC-style int4 vector + address-space-qualified pointers for the async builtin:
// signature is (v4i AS1*, v4i AS3*, imm offset, imm cpol).
typedef int v4i_g __attribute__((vector_size(16)));
typedef __attribute__((address_space(1))) v4i_g* as1_v4i_p;
typedef __attribute__((address_space(3))) v4i_g* as3_v4i_p;

struct U16x16 { v8us lo; v8us hi; };

__device__ __forceinline__ unsigned short f2bf(float f) {
  unsigned int u = __float_as_uint(f);
  u += 0x7fffu + ((u >> 16) & 1u);   // round-to-nearest-even
  return (unsigned short)(u >> 16);
}

__device__ __forceinline__ v8f v8f_zero() {
  v8f z;
#pragma unroll
  for (int i = 0; i < 8; ++i) z[i] = 0.f;
  return z;
}

// A fragment (16x32 bf16, MxK): lane m = lane&15, khalf = lane>>4.
// VGPRs 0-3 hold K = khalf*8 .. +7 ; VGPRs 4-7 hold K = 16+khalf*8 .. +7.
__device__ __forceinline__ v16bf frag_a(const unsigned short* T, int m, int khalf) {
  const unsigned short* p = T + m * TS_STR + khalf * 8;
  U16x16 t;
  t.lo = *(const v8us*)p;
  t.hi = *(const v8us*)(p + 16);
  return __builtin_bit_cast(v16bf, t);
}

// B fragment (32x16 bf16, KxN), LDS stored N-major [BN][BK]:
// lane n = lane&15, VGPRs 0-3 hold K = khalf*8 .. +7 ; VGPRs 4-7 hold K = 16+khalf*8 .. +7.
__device__ __forceinline__ v16bf frag_b(const unsigned short* T, int n, int khalf) {
  const unsigned short* p = T + n * TS_STR + khalf * 8;
  U16x16 t;
  t.lo = *(const v8us*)p;
  t.hi = *(const v8us*)(p + 16);
  return __builtin_bit_cast(v16bf, t);
}

__device__ __forceinline__ v8f wmma_bf16(v16bf a, v16bf b, v8f c) {
  return __builtin_amdgcn_wmma_f32_16x16x32_bf16(false, a, false, b, (short)0, c,
                                                 false, false);
}

// 16-byte global -> LDS stage; async (ASYNCcnt) when available.
__device__ __forceinline__ void stage_b128(const unsigned short* g, unsigned short* l) {
#ifdef ASYNC_LDS
  __builtin_amdgcn_global_load_async_to_lds_b128(
      (as1_v4i_p)(void*)g, (as3_v4i_p)(void*)l, 0, 0);
#else
  *(v8us*)l = *(const v8us*)g;
#endif
}

__device__ __forceinline__ void wait_stage() {
#ifdef ASYNC_LDS
#if __has_builtin(__builtin_amdgcn_s_wait_asynccnt)
  __builtin_amdgcn_s_wait_asynccnt(0);
#else
  asm volatile("s_wait_asynccnt 0x0" ::: "memory");
#endif
#endif
}

__device__ __forceinline__ float silu_f(float x) { return x / (1.f + __expf(-x)); }

// ---------------- fp32 -> bf16 conversion (row-major copy) ----------------
__global__ __launch_bounds__(256)
void k_cvt_bf16(const float* __restrict__ src, unsigned short* __restrict__ dst,
                long long n4) {
  long long i = (long long)blockIdx.x * 256 + threadIdx.x;
  if (i >= n4) return;
  float4 v = ((const float4*)src)[i];
  v4us o;
  o[0] = f2bf(v.x); o[1] = f2bf(v.y); o[2] = f2bf(v.z); o[3] = f2bf(v.w);
  ((v4us*)dst)[i] = o;
}

// ---------------- fp32 [R][C] -> bf16 transposed [C][R] (batched) ----------
__global__ __launch_bounds__(256)
void k_cvt_bf16_t(const float* __restrict__ src, unsigned short* __restrict__ dst,
                  int R, int C) {
  __shared__ float tile[32][33];
  const size_t bo = (size_t)blockIdx.z * R * C;
  src += bo; dst += bo;
  const int r0 = blockIdx.y * 32;
  const int c0 = blockIdx.x * 32;
  const int tx = threadIdx.x & 31;
  const int ty = threadIdx.x >> 5;  // 0..7
#pragma unroll
  for (int i = 0; i < 32; i += 8)
    tile[ty + i][tx] = src[(size_t)(r0 + ty + i) * C + c0 + tx];
  __syncthreads();
#pragma unroll
  for (int i = 0; i < 32; i += 8)
    dst[(size_t)(c0 + ty + i) * R + r0 + tx] = f2bf(tile[tx][ty + i]);
}

__global__ void k_zero_cnt(int* cnt) {
  if (threadIdx.x < E_NUM) cnt[threadIdx.x] = 0;
}

// ---------------- router: one wave32 per token ----------------
__global__ __launch_bounds__(256)
void k_router(const float* __restrict__ x, const float* __restrict__ rw,
              int* __restrict__ cnt, int* __restrict__ tokList,
              float* __restrict__ probList, int N) {
  const int lane = threadIdx.x & 31;
  const int wid = threadIdx.x >> 5;
  const int t = blockIdx.x * 8 + wid;
  if (t >= N) return;

  float s[E_NUM];
#pragma unroll
  for (int e = 0; e < E_NUM; ++e) s[e] = 0.f;

  for (int d = lane; d < D_DIM; d += 32) {
    float xv = x[(size_t)t * D_DIM + d];
    const float* r = rw + (size_t)d * E_NUM;
#pragma unroll
    for (int e = 0; e < E_NUM; ++e) s[e] += xv * r[e];
  }
#pragma unroll
  for (int e = 0; e < E_NUM; ++e) {
#pragma unroll
    for (int off = 16; off > 0; off >>= 1)
      s[e] += __shfl_xor(s[e], off, 32);
  }
  if (lane == 0) {
    float mx = s[0];
#pragma unroll
    for (int e = 1; e < E_NUM; ++e) mx = fmaxf(mx, s[e]);
    float pe[E_NUM], sum = 0.f;
#pragma unroll
    for (int e = 0; e < E_NUM; ++e) { pe[e] = __expf(s[e] - mx); sum += pe[e]; }
#pragma unroll
    for (int e = 0; e < E_NUM; ++e) pe[e] /= sum;
    int i0 = 0;
#pragma unroll
    for (int e = 1; e < E_NUM; ++e) if (pe[e] > pe[i0]) i0 = e;
    int i1 = (i0 == 0) ? 1 : 0;
#pragma unroll
    for (int e = 0; e < E_NUM; ++e)
      if (e != i0 && pe[e] > pe[i1]) i1 = e;
    float denom = pe[i0] + pe[i1] + 1e-9f;
    float p0 = pe[i0] / denom, p1 = pe[i1] / denom;
    int s0 = atomicAdd(&cnt[i0], 1);
    tokList[i0 * N + s0] = t; probList[i0 * N + s0] = p0;
    int s1 = atomicAdd(&cnt[i1], 1);
    tokList[i1 * N + s1] = t; probList[i1 * N + s1] = p1;
  }
}

__global__ void k_prefix(const int* __restrict__ cnt, int* __restrict__ base) {
  int acc = 0;
#pragma unroll
  for (int e = 0; e < E_NUM; ++e) { base[e] = acc; acc += cnt[e]; }
}

// ---------------- fused gate+up GEMM + SiLU (double-buffered LDS) ----------
// Weights are pre-transposed: wT layout [F][D] -> B tile staging == A staging.
template <bool GATHER>
__global__ __launch_bounds__(256)
void k_gateup(const unsigned short* __restrict__ xb,
              const unsigned short* __restrict__ wgT,
              const unsigned short* __restrict__ wuT,
              unsigned short* __restrict__ hOut,
              const int* __restrict__ cnt, const int* __restrict__ base,
              const int* __restrict__ tokList, int N) {
  __shared__ unsigned short As[2][BM * TS_STR];
  __shared__ unsigned short Bg[2][BN * TS_STR];
  __shared__ unsigned short Bu[2][BN * TS_STR];

  int e = 0, rowsHere = N, rbase, hRowBase = 0;
  const unsigned short *gW, *uW;
  if constexpr (GATHER) {
    const int tilesPerE = N / BM;
    e = blockIdx.x / tilesPerE;
    int rt = blockIdx.x - e * tilesPerE;
    rowsHere = cnt[e];
    rbase = rt * BM;
    if (rbase >= rowsHere) return;
    hRowBase = base[e];
    gW = wgT + (size_t)e * D_DIM * F_DIM;
    uW = wuT + (size_t)e * D_DIM * F_DIM;
  } else {
    rbase = blockIdx.x * BM;
    gW = wgT; uW = wuT;
  }
  const int cbase = blockIdx.y * BN;
  const int tid = threadIdx.x;
  const int lane = tid & 31;
  const int wid = tid >> 5;
  const int wm = wid & 3;   // 4 waves over M (32 rows each)
  const int wn = wid >> 2;  // 2 waves over N (64 cols each)
  const int ln = lane & 15;
  const int khalf = lane >> 4;

  v8f accg[2][4], accu[2][4];
#pragma unroll
  for (int im = 0; im < 2; ++im)
#pragma unroll
    for (int jn = 0; jn < 4; ++jn) { accg[im][jn] = v8f_zero(); accu[im][jn] = v8f_zero(); }

  // per-thread chunk mapping (same for A and B tiles: 128 rows x 32 cols)
  int rowc[2], colc[2];
  const unsigned short* aRow[2];
#pragma unroll
  for (int i = 0; i < 2; ++i) {
    int c = tid + i * 256;
    rowc[i] = c >> 2;
    colc[i] = (c & 3) * 8;
    if constexpr (GATHER) {
      int gr = rbase + rowc[i];
      int grc = (gr < rowsHere) ? gr : (rowsHere - 1);   // clamp; masked later
      aRow[i] = xb + (size_t)tokList[e * N + grc] * D_DIM;
    } else {
      aRow[i] = xb + (size_t)(rbase + rowc[i]) * D_DIM;
    }
  }

  auto stage = [&](int kk, int buf) {
#pragma unroll
    for (int i = 0; i < 2; ++i) {
      int lo = rowc[i] * TS_STR + colc[i];
      stage_b128(aRow[i] + kk + colc[i], &As[buf][lo]);
      stage_b128(gW + (size_t)(cbase + rowc[i]) * D_DIM + kk + colc[i], &Bg[buf][lo]);
      stage_b128(uW + (size_t)(cbase + rowc[i]) * D_DIM + kk + colc[i], &Bu[buf][lo]);
    }
  };

  stage(0, 0);
  wait_stage();
  __syncthreads();

  const int KT = D_DIM / BK;
  for (int t = 0; t < KT; ++t) {
    const int cur = t & 1;
    if (t + 1 < KT) stage((t + 1) * BK, cur ^ 1);

    v16bf a0 = frag_a(&As[cur][0], wm * 32 + ln, khalf);
    v16bf a1 = frag_a(&As[cur][0], wm * 32 + 16 + ln, khalf);
#pragma unroll
    for (int jn = 0; jn < 4; ++jn) {
      int n = wn * 64 + jn * 16 + ln;
      v16bf bg = frag_b(&Bg[cur][0], n, khalf);
      v16bf bu = frag_b(&Bu[cur][0], n, khalf);
      accg[0][jn] = wmma_bf16(a0, bg, accg[0][jn]);
      accg[1][jn] = wmma_bf16(a1, bg, accg[1][jn]);
      accu[0][jn] = wmma_bf16(a0, bu, accu[0][jn]);
      accu[1][jn] = wmma_bf16(a1, bu, accu[1][jn]);
    }
    wait_stage();
    __syncthreads();
  }

#pragma unroll
  for (int im = 0; im < 2; ++im) {
#pragma unroll
    for (int jn = 0; jn < 4; ++jn) {
      int n = cbase + wn * 64 + jn * 16 + ln;
#pragma unroll
      for (int r = 0; r < 8; ++r) {
        int mLoc = wm * 32 + im * 16 + khalf * 8 + r;
        float h = silu_f(accg[im][jn][r]) * accu[im][jn][r];
        if constexpr (GATHER) {
          int lr = rbase + mLoc;
          if (lr < rowsHere)
            hOut[(size_t)(hRowBase + lr) * F_DIM + n] = f2bf(h);
        } else {
          hOut[(size_t)(rbase + mLoc) * F_DIM + n] = f2bf(h);
        }
      }
    }
  }
}

// ---------------- down projection GEMM (double-buffered LDS) ---------------
// Weights pre-transposed: wT layout [D][F] (N-major over output dim D).
template <bool ROUTED>
__global__ __launch_bounds__(256)
void k_down(const unsigned short* __restrict__ hIn,
            const unsigned short* __restrict__ wT,
            float* __restrict__ out,
            const int* __restrict__ cnt, const int* __restrict__ base,
            const int* __restrict__ tokList, const float* __restrict__ probList,
            int N) {
  __shared__ unsigned short As[2][BM * TS_STR];
  __shared__ unsigned short Bt[2][BN * TS_STR];

  int e = 0, rowsHere = N, rbase, hRowBase = 0;
  const unsigned short* W;
  if constexpr (ROUTED) {
    const int tilesPerE = N / BM;
    e = blockIdx.x / tilesPerE;
    int rt = blockIdx.x - e * tilesPerE;
    rowsHere = cnt[e];
    rbase = rt * BM;
    if (rbase >= rowsHere) return;
    hRowBase = base[e];
    W = wT + (size_t)e * F_DIM * D_DIM;
  } else {
    rbase = blockIdx.x * BM;
    W = wT;
  }
  const int cbase = blockIdx.y * BN;
  const int tid = threadIdx.x;
  const int lane = tid & 31;
  const int wid = tid >> 5;
  const int wm = wid & 3;
  const int wn = wid >> 2;
  const int ln = lane & 15;
  const int khalf = lane >> 4;

  v8f acc[2][4];
#pragma unroll
  for (int im = 0; im < 2; ++im)
#pragma unroll
    for (int jn = 0; jn < 4; ++jn) acc[im][jn] = v8f_zero();

  int rowc[2], colc[2];
#pragma unroll
  for (int i = 0; i < 2; ++i) {
    int c = tid + i * 256;
    rowc[i] = c >> 2;
    colc[i] = (c & 3) * 8;
  }

  auto stage = [&](int kk, int buf) {
#pragma unroll
    for (int i = 0; i < 2; ++i) {
      int lo = rowc[i] * TS_STR + colc[i];
      stage_b128(hIn + (size_t)(hRowBase + rbase + rowc[i]) * F_DIM + kk + colc[i],
                 &As[buf][lo]);
      stage_b128(W + (size_t)(cbase + rowc[i]) * F_DIM + kk + colc[i], &Bt[buf][lo]);
    }
  };

  stage(0, 0);
  wait_stage();
  __syncthreads();

  const int KT = F_DIM / BK;
  for (int t = 0; t < KT; ++t) {
    const int cur = t & 1;
    if (t + 1 < KT) stage((t + 1) * BK, cur ^ 1);

    v16bf a0 = frag_a(&As[cur][0], wm * 32 + ln, khalf);
    v16bf a1 = frag_a(&As[cur][0], wm * 32 + 16 + ln, khalf);
#pragma unroll
    for (int jn = 0; jn < 4; ++jn) {
      int n = wn * 64 + jn * 16 + ln;
      v16bf b = frag_b(&Bt[cur][0], n, khalf);
      acc[0][jn] = wmma_bf16(a0, b, acc[0][jn]);
      acc[1][jn] = wmma_bf16(a1, b, acc[1][jn]);
    }
    wait_stage();
    __syncthreads();
  }

#pragma unroll
  for (int im = 0; im < 2; ++im) {
#pragma unroll
    for (int jn = 0; jn < 4; ++jn) {
      int n = cbase + wn * 64 + jn * 16 + ln;
#pragma unroll
      for (int r = 0; r < 8; ++r) {
        int mLoc = wm * 32 + im * 16 + khalf * 8 + r;
        float val = acc[im][jn][r];
        if constexpr (ROUTED) {
          int lr = rbase + mLoc;
          if (lr < rowsHere) {
            int tok = tokList[e * N + lr];
            float pw = probList[e * N + lr];
            atomicAdd(&out[(size_t)tok * D_DIM + n], pw * val);
          }
        } else {
          out[(size_t)(rbase + mLoc) * D_DIM + n] = val;
        }
      }
    }
  }
}

extern "C" void kernel_launch(void* const* d_in, const int* in_sizes, int n_in,
                              void* d_out, int out_size, void* d_ws, size_t ws_size,
                              hipStream_t stream) {
  (void)n_in; (void)out_size; (void)ws_size;
  const float* x  = (const float*)d_in[0];
  const float* rw = (const float*)d_in[1];
  const float* sg = (const float*)d_in[2];
  const float* su = (const float*)d_in[3];
  const float* sd = (const float*)d_in[4];
  const float* wg = (const float*)d_in[5];
  const float* wu = (const float*)d_in[6];
  const float* wd = (const float*)d_in[7];
  float* out = (float*)d_out;
  const int N = in_sizes[0] / D_DIM;  // 8192 tokens

  char* p = (char*)d_ws;
  size_t off = 0;
  auto take = [&](size_t bytes) {
    char* r = p + off;
    off = (off + bytes + 255) & ~(size_t)255;
    return r;
  };
  unsigned short* xb   = (unsigned short*)take((size_t)N * D_DIM * 2);
  unsigned short* sgT  = (unsigned short*)take((size_t)D_DIM * F_DIM * 2);
  unsigned short* suT  = (unsigned short*)take((size_t)D_DIM * F_DIM * 2);
  unsigned short* sdT  = (unsigned short*)take((size_t)F_DIM * D_DIM * 2);
  unsigned short* wgT  = (unsigned short*)take((size_t)E_NUM * D_DIM * F_DIM * 2);
  unsigned short* wuT  = (unsigned short*)take((size_t)E_NUM * D_DIM * F_DIM * 2);
  unsigned short* wdT  = (unsigned short*)take((size_t)E_NUM * F_DIM * D_DIM * 2);
  unsigned short* hs   = (unsigned short*)take((size_t)N * F_DIM * 2);
  unsigned short* hr   = (unsigned short*)take((size_t)(N * 2 + BM) * F_DIM * 2);
  int*   tokList  = (int*)take((size_t)E_NUM * N * sizeof(int));
  float* probList = (float*)take((size_t)E_NUM * N * sizeof(float));
  int*   cnt      = (int*)take(256);
  int*   basep    = (int*)take(256);

  k_zero_cnt<<<1, 32, 0, stream>>>(cnt);

  {  // x: plain row-major convert (consumed as A)
    long long n4 = ((long long)N * D_DIM) / 4;
    k_cvt_bf16<<<(int)((n4 + 255) / 256), 256, 0, stream>>>(x, xb, n4);
  }
  // weights: convert + transpose to N-major
  k_cvt_bf16_t<<<dim3(F_DIM / 32, D_DIM / 32, 1), 256, 0, stream>>>(sg, sgT, D_DIM, F_DIM);
  k_cvt_bf16_t<<<dim3(F_DIM / 32, D_DIM / 32, 1), 256, 0, stream>>>(su, suT, D_DIM, F_DIM);
  k_cvt_bf16_t<<<dim3(D_DIM / 32, F_DIM / 32, 1), 256, 0, stream>>>(sd, sdT, F_DIM, D_DIM);
  k_cvt_bf16_t<<<dim3(F_DIM / 32, D_DIM / 32, E_NUM), 256, 0, stream>>>(wg, wgT, D_DIM, F_DIM);
  k_cvt_bf16_t<<<dim3(F_DIM / 32, D_DIM / 32, E_NUM), 256, 0, stream>>>(wu, wuT, D_DIM, F_DIM);
  k_cvt_bf16_t<<<dim3(D_DIM / 32, F_DIM / 32, E_NUM), 256, 0, stream>>>(wd, wdT, F_DIM, D_DIM);

  k_router<<<N / 8, 256, 0, stream>>>(x, rw, cnt, tokList, probList, N);
  k_prefix<<<1, 1, 0, stream>>>(cnt, basep);

  dim3 gS(N / BM, F_DIM / BN);
  k_gateup<false><<<gS, 256, 0, stream>>>(xb, sgT, suT, hs, nullptr, nullptr, nullptr, N);

  dim3 gSD(N / BM, D_DIM / BN);
  k_down<false><<<gSD, 256, 0, stream>>>(hs, sdT, out, nullptr, nullptr, nullptr, nullptr, N);

  dim3 gR(E_NUM * (N / BM), F_DIM / BN);
  k_gateup<true><<<gR, 256, 0, stream>>>(xb, wgT, wuT, hr, cnt, basep, tokList, N);

  dim3 gRD(E_NUM * (N / BM), D_DIM / BN);
  k_down<true><<<gRD, 256, 0, stream>>>(hr, wdT, out, cnt, basep, tokList, probList, N);
}
